// IPAttnProcessor2_0_23613730193503
// MI455X (gfx1250) — compile-verified, matrix operand order checked by reading, not run on hardware
//
#include <hip/hip_runtime.h>
#include <hip/hip_bf16.h>

typedef __attribute__((ext_vector_type(16))) _Float16 v16h;
typedef __attribute__((ext_vector_type(8)))  float    v8f;

#define HEADS 20
#define HDIM  64
#define BATCH 8
#define SEQ   4096
#define TXT   77
#define NIPT  4
#define HID   1280

union FragH { v16h v; unsigned int u[8]; };

__device__ __forceinline__ v8f wmma16(v16h a, v16h b, v8f c) {
  return __builtin_amdgcn_wmma_f32_16x16x32_f16(false, a, false, b, (short)0, c,
                                                false, false);
}

// Async DMA: 16 bytes global -> LDS, tracked by ASYNCcnt.
// LDS operand = low 32 bits of the generic pointer.
__device__ __forceinline__ void async_load_b128(_Float16* lds_dst, const _Float16* g_src) {
  unsigned l = (unsigned)(unsigned long long)(void*)lds_dst;
  asm volatile("global_load_async_to_lds_b128 %0, %1, off"
               :: "v"(l), "v"(g_src) : "memory");
}
#define WAIT_ASYNC(n) asm volatile("s_wait_asynccnt " #n ::: "memory")

// Load a 16x32 (rows x K) fragment in WMMA A/B register layout from a
// row-major [row][k] tile. row = lane&15 (M for A, N for B); per ISA table,
// VGPR v holds f16 pair at K = 16*(v/4) + 2*(v%4) + 8*(lane/16).
__device__ __forceinline__ v16h load_frag(const _Float16* base, int ldm, int lane, int k0) {
  const int row = lane & 15;
  const int hf  = lane >> 4;
  FragH fr;
  const unsigned int* p = (const unsigned int*)(base + row * ldm + k0);
#pragma unroll
  for (int r = 0; r < 8; ++r) {
    int kk = (r >> 2) * 16 + (r & 3) * 2 + hf * 8;
    fr.u[r] = p[kk >> 1];
  }
  return fr.v;
}

__device__ __forceinline__ v16h load_frag_g(const _Float16* base, size_t ldm, int lane, int k0) {
  const int row = lane & 15;
  const int hf  = lane >> 4;
  FragH fr;
  const unsigned int* p = (const unsigned int*)(base + (size_t)row * ldm + k0);
#pragma unroll
  for (int r = 0; r < 8; ++r) {
    int kk = (r >> 2) * 16 + (r & 3) * 2 + hf * 8;
    fr.u[r] = p[kk >> 1];
  }
  return fr.v;
}

// ---------------------------------------------------------------------------
// Weight pre-pass: W f32 [K][N] -> Wt f16 [N][K] (transposed, DMA-friendly).
// Grid (K/32, N/32), 256 threads.
// ---------------------------------------------------------------------------
__global__ __launch_bounds__(256) void transpose_convert_kernel(
    const float* __restrict__ W, _Float16* __restrict__ Wt, int K, int N) {
  __shared__ float tile[32][33];
  const int k0 = blockIdx.x * 32, n0 = blockIdx.y * 32;
  const int tx = threadIdx.x & 31, ty = threadIdx.x >> 5;  // ty 0..7
#pragma unroll
  for (int i = ty; i < 32; i += 8)
    tile[i][tx] = W[(size_t)(k0 + i) * N + n0 + tx];
  __syncthreads();
#pragma unroll
  for (int i = ty; i < 32; i += 8)
    Wt[(size_t)(n0 + i) * K + k0 + tx] = (_Float16)tile[tx][i];
}

// ---------------------------------------------------------------------------
// WMMA GEMM: C[M,N] = A[M,K] @ Wt^T (+ bias), Wt is f16 [N][K].
// 256 threads = 8 waves; block tile 128x128, K-step 32, double-buffered LDS,
// async-DMA staging for f16 operands. A_HALF path requires M % 128 == 0.
// ---------------------------------------------------------------------------
template <bool A_HALF, bool OUT_HALF, bool BIAS>
__global__ __launch_bounds__(256) void wmma_gemm_kernel(
    const void* __restrict__ Ain, const _Float16* __restrict__ Wt,
    const float* __restrict__ bias, void* __restrict__ Cout,
    int M, int K, int N) {
  __shared__ __align__(16) _Float16 sA[2][128 * 32];  // [m][k]
  __shared__ __align__(16) _Float16 sB[2][128 * 32];  // [n][k]

  const float*    Af = (const float*)Ain;
  const _Float16* Ah = (const _Float16*)Ain;

  const int tid  = threadIdx.x;
  const int lane = tid & 31;
  const int w    = tid >> 5;
  const int m0   = blockIdx.x * 128;
  const int n0   = blockIdx.y * 128;

  auto stage = [&](int buf, int k0) {
    // weights: async DMA, [n][k] contiguous rows (2 instr/thread)
#pragma unroll
    for (int it = 0; it < 2; ++it) {
      int f   = tid + it * 256;   // 0..511
      int row = f >> 2;           // 0..127
      int seg = (f & 3) << 3;     // halfs: 0,8,16,24
      async_load_b128(&sB[buf][row * 32 + seg],
                      Wt + (size_t)(n0 + row) * K + k0 + seg);
    }
    if (A_HALF) {
      // f16 activations: async DMA (2 instr/thread); M % 128 == 0 here.
#pragma unroll
      for (int it = 0; it < 2; ++it) {
        int f   = tid + it * 256;
        int row = f >> 2;
        int seg = (f & 3) << 3;
        async_load_b128(&sA[buf][row * 32 + seg],
                        Ah + (size_t)(m0 + row) * K + k0 + seg);
      }
    } else {
      // f32 activations: load + convert (guarded for M tails)
#pragma unroll
      for (int it = 0; it < 4; ++it) {
        int f   = tid + it * 256;  // 1024 quads
        int row = f >> 3;
        int kp  = (f & 7) << 2;
        int gr  = m0 + row;
        float4 a4 = make_float4(0.f, 0.f, 0.f, 0.f);
        if (gr < M) a4 = *(const float4*)(Af + (size_t)gr * K + k0 + kp);
        sA[buf][row * 32 + kp + 0] = (_Float16)a4.x;
        sA[buf][row * 32 + kp + 1] = (_Float16)a4.y;
        sA[buf][row * 32 + kp + 2] = (_Float16)a4.z;
        sA[buf][row * 32 + kp + 3] = (_Float16)a4.w;
      }
      if (k0 + 32 < K) {  // prefetch next activation slab
        int gr = m0 + (tid & 127);
        if (gr < M) __builtin_prefetch(Af + (size_t)gr * K + k0 + 32, 0, 0);
      }
    }
  };

  v8f acc[8] = {};
  stage(0, 0);
  int cur = 0;
  for (int k0 = 0; k0 < K; k0 += 32) {
    if (k0 + 32 < K) {
      stage(cur ^ 1, k0 + 32);   // overlap next-tile DMA with this tile's math
      if (A_HALF) { WAIT_ASYNC(0x4); } else { WAIT_ASYNC(0x2); }  // cur tile done
    } else {
      WAIT_ASYNC(0x0);
    }
    __syncthreads();  // all waves' cur-tile staging visible

    v16h af = load_frag(&sA[cur][(w * 16) * 32], 32, lane, 0);
#pragma unroll
    for (int nt = 0; nt < 8; ++nt) {
      v16h bf = load_frag(&sB[cur][(nt * 16) * 32], 32, lane, 0);
      acc[nt] = wmma16(af, bf, acc[nt]);
    }
    __syncthreads();  // done reading cur before it is restaged
    cur ^= 1;
  }

  // epilogue: C layout row m = r + 8*(lane/16), col n = lane&15
  const int hf = lane >> 4;
#pragma unroll
  for (int nt = 0; nt < 8; ++nt) {
#pragma unroll
    for (int r = 0; r < 8; ++r) {
      int m = m0 + w * 16 + r + 8 * hf;
      int n = n0 + nt * 16 + (lane & 15);
      if (m < M) {
        float val = acc[nt][r];
        if (BIAS) val += bias[n];
        if (OUT_HALF) ((_Float16*)Cout)[(size_t)m * N + n] = (_Float16)val;
        else          ((float*)Cout)[(size_t)m * N + n]    = val;
      }
    }
  }
}

// ---------------------------------------------------------------------------
// Fused dual-softmax attention: txt = softmax(QK^T/8)V over 77 keys,
// ip = softmax(QKip^T/8)Vip over 4 keys, combined = txt + ip.
// Grid: (S/128, B*HEADS); 256 threads = 8 waves; wave handles 16 queries.
// ---------------------------------------------------------------------------
__global__ __launch_bounds__(256) void ip_attn_kernel(
    const _Float16* __restrict__ qbuf, const _Float16* __restrict__ kbuf,
    const _Float16* __restrict__ vbuf, const _Float16* __restrict__ kipbuf,
    const _Float16* __restrict__ vipbuf, _Float16* __restrict__ comb) {
  __shared__ __align__(16) _Float16 sK[80 * 64];     // [t][d], t padded to 80
  __shared__ __align__(16) _Float16 sVt[64 * 96];    // [d][t], t padded to 96
  __shared__ __align__(16) _Float16 sKip[16 * 64];   // [t][d]
  __shared__ __align__(16) _Float16 sVipt[64 * 32];  // [d][t]
  __shared__ __align__(16) _Float16 sP[8 * 16 * 96];  // per-wave P tiles
  __shared__ __align__(16) _Float16 sPip[8 * 16 * 32];

  const int tid  = threadIdx.x;
  const int lane = tid & 31;
  const int w    = tid >> 5;
  const int b    = blockIdx.y / HEADS;
  const int h    = blockIdx.y % HEADS;
  const int s0   = blockIdx.x * 128 + w * 16;

  // K / K_ip: layout-preserving rows -> async DMA (16B segments); pad = zeros.
  for (int f = tid; f < 80 * 8; f += 256) {
    int t = f >> 3, seg = (f & 7) << 3;
    if (t < TXT)
      async_load_b128(&sK[t * 64 + seg],
                      kbuf + ((size_t)(b * TXT + t)) * HID + h * HDIM + seg);
    else
      *(uint4*)&sK[t * 64 + seg] = make_uint4(0u, 0u, 0u, 0u);
  }
  for (int f = tid; f < 16 * 8; f += 256) {
    int t = f >> 3, seg = (f & 7) << 3;
    if (t < NIPT)
      async_load_b128(&sKip[t * 64 + seg],
                      kipbuf + ((size_t)(b * NIPT + t)) * HID + h * HDIM + seg);
    else
      *(uint4*)&sKip[t * 64 + seg] = make_uint4(0u, 0u, 0u, 0u);
  }
  // V / V_ip transposed [d][t] (DMA cannot transpose): manual stage.
  for (int i = tid; i < 64 * 96; i += 256) {
    int d = i / 96, t = i % 96;
    sVt[i] = (t < TXT) ? vbuf[((size_t)(b * TXT + t)) * HID + h * HDIM + d] : (_Float16)0.f;
  }
  for (int i = tid; i < 64 * 32; i += 256) {
    int d = i >> 5, t = i & 31;
    sVipt[i] = (t < NIPT) ? vipbuf[((size_t)(b * NIPT + t)) * HID + h * HDIM + d] : (_Float16)0.f;
  }
  for (int i = tid; i < 8 * 16 * 96; i += 256) sP[i]   = (_Float16)0.f;
  for (int i = tid; i < 8 * 16 * 32; i += 256) sPip[i] = (_Float16)0.f;
  WAIT_ASYNC(0x0);
  __syncthreads();

  const _Float16* qbase = qbuf + ((size_t)(b * SEQ + s0)) * HID + h * HDIM;
  v16h qa0 = load_frag_g(qbase, HID, lane, 0);
  v16h qa1 = load_frag_g(qbase, HID, lane, 32);

  // scores
  v8f sc[5] = {};
#pragma unroll
  for (int t = 0; t < 5; ++t) {
    v16h b0 = load_frag(&sK[(t * 16) * 64], 64, lane, 0);
    sc[t] = wmma16(qa0, b0, sc[t]);
    v16h b1 = load_frag(&sK[(t * 16) * 64], 64, lane, 32);
    sc[t] = wmma16(qa1, b1, sc[t]);
  }
  v8f sip = {};
  {
    v16h b0 = load_frag(sKip, 64, lane, 0);
    sip = wmma16(qa0, b0, sip);
    v16h b1 = load_frag(sKip, 64, lane, 32);
    sip = wmma16(qa1, b1, sip);
  }

  const float scale = 0.125f;  // 1/sqrt(64)
  const int tl = lane & 15;
  const int hf = lane >> 4;
  _Float16* sPw  = &sP[w * 16 * 96];
  _Float16* sPiw = &sPip[w * 16 * 32];

#pragma unroll
  for (int r = 0; r < 8; ++r) {
    // text softmax, row m = r + 8*hf; key t = 16*tile + tl, valid t < 77
    float mx = -3.0e38f;
#pragma unroll
    for (int t = 0; t < 5; ++t) {
      float x = sc[t][r] * scale;
      if (t == 4 && tl >= 13) x = -3.0e38f;
      mx = fmaxf(mx, x);
    }
#pragma unroll
    for (int off = 1; off < 16; off <<= 1) mx = fmaxf(mx, __shfl_xor(mx, off, 32));
    float sum = 0.f;
#pragma unroll
    for (int t = 0; t < 5; ++t) {
      float x = sc[t][r] * scale;
      float e = (t == 4 && tl >= 13) ? 0.f : __expf(x - mx);
      sc[t][r] = e;
      sum += e;
    }
#pragma unroll
    for (int off = 1; off < 16; off <<= 1) sum += __shfl_xor(sum, off, 32);
    float inv = 1.0f / sum;
#pragma unroll
    for (int t = 0; t < 5; ++t)
      sPw[(r + 8 * hf) * 96 + t * 16 + tl] = (_Float16)(sc[t][r] * inv);

    // ip softmax (4 valid keys)
    float xi = sip[r] * scale;
    float mi = (tl >= NIPT) ? -3.0e38f : xi;
#pragma unroll
    for (int off = 1; off < 16; off <<= 1) mi = fmaxf(mi, __shfl_xor(mi, off, 32));
    float ei = (tl >= NIPT) ? 0.f : __expf(xi - mi);
    float si = ei;
#pragma unroll
    for (int off = 1; off < 16; off <<= 1) si += __shfl_xor(si, off, 32);
    sPiw[(r + 8 * hf) * 32 + tl] = (_Float16)(ei / si);
  }

  // P @ V (P pre-normalized, so txt and ip just accumulate)
  v8f o[4] = {};
  {
    v16h pa = load_frag(sPiw, 32, lane, 0);
#pragma unroll
    for (int nt = 0; nt < 4; ++nt) {
      v16h vb = load_frag(&sVipt[(nt * 16) * 32], 32, lane, 0);
      o[nt] = wmma16(pa, vb, o[nt]);
    }
  }
#pragma unroll
  for (int ks = 0; ks < 3; ++ks) {
    v16h pa = load_frag(sPw, 96, lane, ks * 32);
#pragma unroll
    for (int nt = 0; nt < 4; ++nt) {
      v16h vb = load_frag(&sVt[(nt * 16) * 96], 96, lane, ks * 32);
      o[nt] = wmma16(pa, vb, o[nt]);
    }
  }

#pragma unroll
  for (int nt = 0; nt < 4; ++nt) {
#pragma unroll
    for (int r = 0; r < 8; ++r) {
      int m = s0 + r + 8 * hf;
      comb[((size_t)(b * SEQ + m)) * HID + h * HDIM + nt * 16 + tl] = (_Float16)o[nt][r];
    }
  }
}

// ---------------------------------------------------------------------------
extern "C" void kernel_launch(void* const* d_in, const int* in_sizes, int n_in,
                              void* d_out, int out_size, void* d_ws, size_t ws_size,
                              hipStream_t stream) {
  (void)in_sizes; (void)n_in; (void)out_size; (void)ws_size;
  const float* hidden = (const float*)d_in[0];   // [8,4096,1280]
  const float* enc    = (const float*)d_in[1];   // [8,77,2048]
  const float* iph    = (const float*)d_in[2];   // [8,4,2048]
  const float* Wq     = (const float*)d_in[3];
  const float* Wk     = (const float*)d_in[4];
  const float* Wv     = (const float*)d_in[5];
  const float* Wkip   = (const float*)d_in[6];
  const float* Wvip   = (const float*)d_in[7];
  const float* Wo     = (const float*)d_in[8];
  const float* bo     = (const float*)d_in[9];
  float* out = (float*)d_out;

  char* ws = (char*)d_ws;
  _Float16* qbuf   = (_Float16*)(ws + 0);           // 32768*1280 f16
  _Float16* comb   = (_Float16*)(ws + 83886080);
  _Float16* kbuf   = (_Float16*)(ws + 167772160);   // 616*1280 f16
  _Float16* vbuf   = (_Float16*)(ws + 169349120);
  _Float16* kipbuf = (_Float16*)(ws + 170926080);   // 32*1280 f16
  _Float16* vipbuf = (_Float16*)(ws + 171008000);
  _Float16* wqt    = (_Float16*)(ws + 171089920);   // 1280*1280 f16 [N][K]
  _Float16* wkt    = (_Float16*)(ws + 174366720);   // 1280*2048 f16
  _Float16* wvt    = (_Float16*)(ws + 179609600);
  _Float16* wkipt  = (_Float16*)(ws + 184852480);
  _Float16* wvipt  = (_Float16*)(ws + 190095360);
  _Float16* wot    = (_Float16*)(ws + 195338240);   // 1280*1280 f16

  dim3 blk(256, 1, 1);

  // Weight pre-pass: f32 [K][N] -> f16 [N][K] for DMA-friendly B tiles.
  transpose_convert_kernel<<<dim3(40, 40), blk, 0, stream>>>(Wq,   wqt,   HID,  HID);
  transpose_convert_kernel<<<dim3(64, 40), blk, 0, stream>>>(Wk,   wkt,   2048, HID);
  transpose_convert_kernel<<<dim3(64, 40), blk, 0, stream>>>(Wv,   wvt,   2048, HID);
  transpose_convert_kernel<<<dim3(64, 40), blk, 0, stream>>>(Wkip, wkipt, 2048, HID);
  transpose_convert_kernel<<<dim3(64, 40), blk, 0, stream>>>(Wvip, wvipt, 2048, HID);
  transpose_convert_kernel<<<dim3(40, 40), blk, 0, stream>>>(Wo,   wot,   HID,  HID);

  // Q projection: [32768,1280] x [1280,1280] -> f16
  wmma_gemm_kernel<false, true, false><<<dim3(256, 10), blk, 0, stream>>>(
      hidden, wqt, nullptr, qbuf, BATCH * SEQ, HID, HID);
  // K/V projections: [616,2048] x [2048,1280] -> f16
  wmma_gemm_kernel<false, true, false><<<dim3(5, 10), blk, 0, stream>>>(
      enc, wkt, nullptr, kbuf, BATCH * TXT, 2048, HID);
  wmma_gemm_kernel<false, true, false><<<dim3(5, 10), blk, 0, stream>>>(
      enc, wvt, nullptr, vbuf, BATCH * TXT, 2048, HID);
  // IP K/V projections: [32,2048] x [2048,1280] -> f16
  wmma_gemm_kernel<false, true, false><<<dim3(1, 10), blk, 0, stream>>>(
      iph, wkipt, nullptr, kipbuf, BATCH * NIPT, 2048, HID);
  wmma_gemm_kernel<false, true, false><<<dim3(1, 10), blk, 0, stream>>>(
      iph, wvipt, nullptr, vipbuf, BATCH * NIPT, 2048, HID);
  // Fused dual-softmax attention -> combined f16 [32768,1280]
  ip_attn_kernel<<<dim3(SEQ / 128, BATCH * HEADS), blk, 0, stream>>>(
      qbuf, kbuf, vbuf, kipbuf, vipbuf, comb);
  // Output projection with bias: [32768,1280] x [1280,1280] + bo -> f32
  wmma_gemm_kernel<true, false, true><<<dim3(256, 10), blk, 0, stream>>>(
      comb, wot, bo, out, BATCH * SEQ, HID, HID);
}
